// cross_update_block_17059610100099
// MI455X (gfx1250) — matrix operand order repaired and checked by reading, not compiled, not run on hardware
//
#include <hip/hip_runtime.h>
#include <math.h>

typedef __attribute__((ext_vector_type(16))) __bf16 v16bf;
typedef __attribute__((ext_vector_type(8)))  __bf16 v8bf;
typedef __attribute__((ext_vector_type(4)))  __bf16 v4bf;
typedef __attribute__((ext_vector_type(8)))  float  v8f;

#define BATCH 16
#define CCH   256
#define NCLS  64
#define SDIM  4096
#define CT    32           // channels per workgroup (2 WMMA M-tiles)
#define SC1   128          // phase-1 S chunk
#define LF    (SC1 + 8)    // padded LDS strides (multiples of 8 -> 16B-aligned vec loads)
#define LP    (SC1 + 8)
#define AT    68           // attn fp32 stride (floats)
#define AB    72           // attn bf16 stride
#define SC2   128          // phase-2 S chunk
#define PT    72           // transposed prior stride (bf16, mult of 8)

__global__ __launch_bounds__(256) void attn_prior_fused_kernel(
    const float* __restrict__ prior,   // [B, NCLS, S]
    const float* __restrict__ feat,    // [B, C, S]
    float* __restrict__ out)           // [B, C, S]
{
    __shared__ __bf16 lds_f[CT * LF];        //  8704 B
    __shared__ __bf16 lds_p[NCLS * LP];      // 17408 B
    __shared__ float  attn_f[CT * AT];       //  8704 B
    __shared__ __bf16 attn_b[CT * AB];       //  4608 B
    __shared__ __bf16 lds_pt[SC2 * PT];      // 18432 B

    const int tid  = threadIdx.x;
    const int wave = tid >> 5;
    const int lane = tid & 31;
    const int half = lane >> 4;      // lane-group select (K/M split per ISA layout)
    const int row  = lane & 15;

    const int b  = blockIdx.x / (CCH / CT);
    const int c0 = (blockIdx.x % (CCH / CT)) * CT;

    const float* fB = feat  + ((size_t)b * CCH + c0) * SDIM;
    const float* pB = prior + (size_t)b * NCLS * SDIM;
    float*       oB = out   + ((size_t)b * CCH + c0) * SDIM;

    const int mi = wave & 1;   // channel sub-tile (M)
    const int ni = wave >> 1;  // class column tile (N) in phase 1

    // ------------------------- Phase 1: attn = f . p^T --------------------------
    v8f acc = {0.f, 0.f, 0.f, 0.f, 0.f, 0.f, 0.f, 0.f};
    for (int s0 = 0; s0 < SDIM; s0 += SC1) {
        if (s0 + SC1 < SDIM) {   // prefetch next chunk (global_prefetch_b8)
            __builtin_prefetch(fB + (size_t)(tid & (CT - 1)) * SDIM + s0 + SC1 + (tid >> 5) * 16, 0, 1);
            __builtin_prefetch(pB + (size_t)(tid & (NCLS - 1)) * SDIM + s0 + SC1 + (tid >> 6) * 32, 0, 1);
        }
        // cooperative fp32->bf16 staging: float4 (b128) global loads, b64 LDS stores
        for (int i = tid; i < CT * SC1 / 4; i += 256) {            // 1024 vec4 -> 4/thread
            int r = i >> 5, cc = (i & 31) << 2;
            float4 v = *(const float4*)&fB[(size_t)r * SDIM + s0 + cc];
            v4bf w = {(__bf16)v.x, (__bf16)v.y, (__bf16)v.z, (__bf16)v.w};
            *(v4bf*)&lds_f[r * LF + cc] = w;
        }
        for (int i = tid; i < NCLS * SC1 / 4; i += 256) {          // 2048 vec4 -> 8/thread
            int r = i >> 5, cc = (i & 31) << 2;
            float4 v = *(const float4*)&pB[(size_t)r * SDIM + s0 + cc];
            v4bf w = {(__bf16)v.x, (__bf16)v.y, (__bf16)v.z, (__bf16)v.w};
            *(v4bf*)&lds_p[r * LP + cc] = w;
        }
        __syncthreads();

        #pragma unroll
        for (int ks = 0; ks < SC1; ks += 32) {
            // A[M=mi*16+row][k]: lanes 0-15 hold K 0-7 & 16-23, lanes 16-31 hold 8-15 & 24-31
            const __bf16* fr = &lds_f[(mi * 16 + row) * LF + ks + 8 * half];
            // B[k][N=ni*16+row] = p[class][s]: lanes 0-15 K 0-7 & 16-23, lanes 16-31 K 8-15 & 24-31
            const __bf16* pr = &lds_p[(ni * 16 + row) * LP + ks + 8 * half];
            v8bf alo = *(const v8bf*)(fr);
            v8bf ahi = *(const v8bf*)(fr + 16);
            v8bf blo = *(const v8bf*)(pr);
            v8bf bhi = *(const v8bf*)(pr + 16);
            v16bf a, bm;
            #pragma unroll
            for (int j = 0; j < 8; ++j) {
                a[j]  = alo[j];  a[8 + j]  = ahi[j];
                bm[j] = blo[j];  bm[8 + j] = bhi[j];
            }
            acc = __builtin_amdgcn_wmma_f32_16x16x32_bf16(
                false, a, false, bm, (short)0, acc, false, false);
        }
        __syncthreads();
    }
    // D layout: VGPR r -> M = r + 8*half, N = row
    #pragma unroll
    for (int r = 0; r < 8; ++r)
        attn_f[(mi * 16 + r + 8 * half) * AT + ni * 16 + row] = acc[r];
    __syncthreads();

    // ----------------------------- softmax over K --------------------------------
    if (tid < CT) {
        float* rp = &attn_f[tid * AT];
        float mx = rp[0];
        for (int k = 1; k < NCLS; ++k) mx = fmaxf(mx, rp[k]);
        float sum = 0.f;
        for (int k = 0; k < NCLS; ++k) {
            float e = __expf((rp[k] - mx) * 0.125f);   // 1/sqrt(64) logit scale
            rp[k] = e; sum += e;
        }
        float inv = 1.0f / sum;
        for (int k = 0; k < NCLS; ++k)
            attn_b[tid * AB + k] = (__bf16)(rp[k] * inv);
    }
    __syncthreads();

    // A operands for phase 2 (K = 64 classes -> two 16x32 A tiles), loaded once
    v16bf a0, a1;
    {
        const __bf16* ar = &attn_b[(mi * 16 + row) * AB];
        v8bf q0 = *(const v8bf*)(ar + 8 * half);
        v8bf q1 = *(const v8bf*)(ar + 16 + 8 * half);
        v8bf q2 = *(const v8bf*)(ar + 32 + 8 * half);
        v8bf q3 = *(const v8bf*)(ar + 48 + 8 * half);
        #pragma unroll
        for (int j = 0; j < 8; ++j) {
            a0[j] = q0[j]; a0[8 + j] = q1[j];
            a1[j] = q2[j]; a1[8 + j] = q3[j];
        }
    }

    // ----------------------- Phase 2: out = feat + attn . p ----------------------
    for (int s0 = 0; s0 < SDIM; s0 += SC2) {
        if (s0 + SC2 < SDIM)
            __builtin_prefetch(pB + (size_t)(tid & (NCLS - 1)) * SDIM + s0 + SC2 + (tid >> 6) * 32, 0, 1);
        // stage prior transposed: lds_pt[s][k]  (B operand wants K contiguous per lane)
        // float4 global reads along s; scalar bf16 LDS writes (transpose)
        for (int i = tid; i < NCLS * SC2 / 4; i += 256) {          // 2048 vec4 -> 8/thread
            int k = i >> 5, ss = (i & 31) << 2;
            float4 v = *(const float4*)&pB[(size_t)k * SDIM + s0 + ss];
            lds_pt[(ss + 0) * PT + k] = (__bf16)v.x;
            lds_pt[(ss + 1) * PT + k] = (__bf16)v.y;
            lds_pt[(ss + 2) * PT + k] = (__bf16)v.z;
            lds_pt[(ss + 3) * PT + k] = (__bf16)v.w;
        }
        __syncthreads();

        for (int st = (wave >> 1); st < SC2 / 16; st += 4) {
            // B[k][N = s-col]: lane (half,row) reads k = 8*half + j and 16 + 8*half + j at s = s0 + st*16 + row
            const __bf16* pr = &lds_pt[(st * 16 + row) * PT + 8 * half];
            v8bf b0l = *(const v8bf*)(pr);
            v8bf b0h = *(const v8bf*)(pr + 16);
            v8bf b1l = *(const v8bf*)(pr + 32);
            v8bf b1h = *(const v8bf*)(pr + 48);
            v16bf b0, b1;
            #pragma unroll
            for (int j = 0; j < 8; ++j) {
                b0[j] = b0l[j]; b0[8 + j] = b0h[j];
                b1[j] = b1l[j]; b1[8 + j] = b1h[j];
            }
            v8f c8 = {0.f, 0.f, 0.f, 0.f, 0.f, 0.f, 0.f, 0.f};
            c8 = __builtin_amdgcn_wmma_f32_16x16x32_bf16(
                false, a0, false, b0, (short)0, c8, false, false);
            c8 = __builtin_amdgcn_wmma_f32_16x16x32_bf16(
                false, a1, false, b1, (short)0, c8, false, false);

            // epilogue: out = feature + c8  (coalesced 64B per half-wave per row);
            // NT stores: output is write-once, keep L2 for the reusable prior
            const size_t off = (size_t)(mi * 16 + 8 * half) * SDIM + s0 + st * 16 + row;
            const float* fp = fB + off;
            float*       op = oB + off;
            #pragma unroll
            for (int r = 0; r < 8; ++r)
                __builtin_nontemporal_store(fp[(size_t)r * SDIM] + c8[r], &op[(size_t)r * SDIM]);
        }
        __syncthreads();
    }
}

extern "C" void kernel_launch(void* const* d_in, const int* in_sizes, int n_in,
                              void* d_out, int out_size, void* d_ws, size_t ws_size,
                              hipStream_t stream) {
    (void)in_sizes; (void)n_in; (void)out_size; (void)d_ws; (void)ws_size;
    const float* prior = (const float*)d_in[0];  // refined_shape_prior [16,64,64,64]
    const float* feat  = (const float*)d_in[1];  // feature             [16,256,64,64]
    float* out = (float*)d_out;                  // float32 output

    dim3 grid(BATCH * (CCH / CT));   // 16 batches x 8 channel tiles = 128 WGs
    dim3 block(256);                 // 8 wave32 waves
    attn_prior_fused_kernel<<<grid, block, 0, stream>>>(prior, feat, out);
}
